// MoCTopKExperts_31336081391816
// MI455X (gfx1250) — matrix operand order repaired and checked by reading, not compiled, not run on hardware
//
#include <hip/hip_runtime.h>
#include <hip/hip_bf16.h>

typedef __bf16 bf16_t;
typedef __attribute__((ext_vector_type(16))) __bf16 v16bf;
typedef __attribute__((ext_vector_type(8)))  __bf16 v8bf;
typedef __attribute__((ext_vector_type(8)))  float  v8f;

#define DEV_INLINE __device__ __forceinline__

constexpr int Bc = 4, Tc = 2048, Dc = 768, Ec = 8, Kc = 2, Hc = 2048;
constexpr int NTOK = Bc * Tc;        // 8192 tokens
constexpr int NKs  = NTOK * Kc;      // 16384 slots
constexpr int CAP  = 4096;           // per-expert capacity (matches reference)
constexpr int D2   = 2 * Dc;         // 1536

// ---------- small helpers ----------
DEV_INLINE bf16_t tobf(float f) {
    unsigned u = __float_as_uint(f);
    unsigned r = (u + 0x7FFFu + ((u >> 16) & 1u)) >> 16;
    return __builtin_bit_cast(bf16_t, (unsigned short)r);
}
DEV_INLINE v8f vzero() {
    v8f v;
    #pragma unroll
    for (int i = 0; i < 8; ++i) v[i] = 0.0f;
    return v;
}

// ---------- software-pipelined WMMA inner loop (ping-pong, unroll x2) ----------
// C[16 x (NT*16)] += A[16 x K] * B[(NT*16) x K]^T   (A,B bf16 row-major, K contiguous)
// Fragment layouts per CDNA5 ISA 7.12.2 (16-bit A 16x32; B 32x16).
// Requires K_ % 64 == 0 (true for all call sites: 768 / 1536 / 2048).
// Two fragment sets alternate roles across the unrolled body, so the compiler
// needs no v_mov rotation and every v_wmma waits only on loads issued a full
// 4-WMMA stage earlier.
template <int NT>
DEV_INLINE void gemm_body(const bf16_t* __restrict__ Arow, int lda,
                          const bf16_t* __restrict__ Brow, int ldb,
                          int K_, int lane, v8f acc[NT]) {
    const int half = lane >> 4;     // 0: lanes 0-15, 1: lanes 16-31
    const int r    = lane & 15;
    const bf16_t* aBase = Arow + (size_t)r * lda + half * 8;
    const bf16_t* bBase = Brow + (size_t)r * ldb + half * 8;

    auto loadA = [&](int kb) -> v16bf {
        v8bf a0 = *(const v8bf*)(aBase + kb);
        v8bf a1 = *(const v8bf*)(aBase + kb + 16);
        v16bf a;
        #pragma unroll
        for (int i = 0; i < 8; ++i) { a[i] = a0[i]; a[i + 8] = a1[i]; }
        return a;
    };
    auto loadB = [&](int t, int kb) -> v16bf {
        const bf16_t* bp = bBase + (size_t)t * 16 * ldb + kb;
        v8bf b0 = *(const v8bf*)bp;
        v8bf b1 = *(const v8bf*)(bp + 16);
        v16bf b;
        #pragma unroll
        for (int i = 0; i < 8; ++i) { b[i] = b0[i]; b[i + 8] = b1[i]; }
        return b;
    };

    // Prologue: stage P <- kb=0
    v16bf aP = loadA(0);
    v16bf bP[NT];
    #pragma unroll
    for (int t = 0; t < NT; ++t) bP[t] = loadB(t, 0);

    for (int kb = 0; kb + 64 <= K_; kb += 64) {
        // Stage Q <- kb+32 (issued before consuming P)
        __builtin_prefetch(aBase + ((kb + 256 < K_) ? (kb + 256) : kb), 0, 1);
        v16bf aQ = loadA(kb + 32);
        v16bf bQ[NT];
        #pragma unroll
        for (int t = 0; t < NT; ++t) bQ[t] = loadB(t, kb + 32);
        #pragma unroll
        for (int t = 0; t < NT; ++t)
            acc[t] = __builtin_amdgcn_wmma_f32_16x16x32_bf16(
                false, aP, false, bP[t], (short)0, acc[t], false, false);

        // Stage P <- kb+64 (wrap to 0 on last trip: redundant, harmless)
        const int kn = (kb + 64 < K_) ? (kb + 64) : 0;
        aP = loadA(kn);
        #pragma unroll
        for (int t = 0; t < NT; ++t) bP[t] = loadB(t, kn);
        #pragma unroll
        for (int t = 0; t < NT; ++t)
            acc[t] = __builtin_amdgcn_wmma_f32_16x16x32_bf16(
                false, aQ, false, bQ[t], (short)0, acc[t], false, false);
    }
}

// ---------- fill / convert ----------
__global__ void fill_f32_kernel(float* p, float v, long long n) {
    long long i = (long long)blockIdx.x * blockDim.x + threadIdx.x;
    long long stride = (long long)gridDim.x * blockDim.x;
    for (; i < n; i += stride) p[i] = v;
}
__global__ void fill_i32_kernel(int* p, int v, long long n) {
    long long i = (long long)blockIdx.x * blockDim.x + threadIdx.x;
    long long stride = (long long)gridDim.x * blockDim.x;
    for (; i < n; i += stride) p[i] = v;
}
__global__ void cvt_bf16_kernel(const float* __restrict__ src, bf16_t* __restrict__ dst,
                                long long n) {
    long long i = (long long)blockIdx.x * blockDim.x + threadIdx.x;
    long long stride = (long long)gridDim.x * blockDim.x;
    for (; i < n; i += stride) dst[i] = tobf(src[i]);
}

// ---------- router: one wave32 per token ----------
__global__ __launch_bounds__(256) void router_kernel(
    const float* __restrict__ xf, const float* __restrict__ gw,
    float* __restrict__ topk_probs, int* __restrict__ topk_idx,
    int* __restrict__ idx_out /* int64 as int pairs */,
    float* __restrict__ rp_sum, float* __restrict__ assign_sum,
    float* __restrict__ zz_sum) {
    int tok = (int)((blockIdx.x * (long long)blockDim.x + threadIdx.x) >> 5);
    if (tok >= NTOK) return;
    int lane = threadIdx.x & 31;
    float acc[Ec];
    #pragma unroll
    for (int e = 0; e < Ec; ++e) acc[e] = 0.0f;
    const float* xr = xf + (size_t)tok * Dc;
    for (int d = lane; d < Dc; d += 32) {
        float xv = xr[d];
        #pragma unroll
        for (int e = 0; e < Ec; ++e) acc[e] += xv * gw[e * Dc + d];
    }
    #pragma unroll
    for (int e = 0; e < Ec; ++e)
        for (int off = 16; off > 0; off >>= 1) acc[e] += __shfl_xor(acc[e], off, 32);
    if (lane == 0) {
        int i0 = 0;
        #pragma unroll
        for (int e = 1; e < Ec; ++e) if (acc[e] > acc[i0]) i0 = e;
        int i1 = (i0 == 0) ? 1 : 0;
        #pragma unroll
        for (int e = 0; e < Ec; ++e) if (e != i0 && acc[e] > acc[i1]) i1 = e;
        float v0 = acc[i0], v1 = acc[i1];
        float p1 = expf(v1 - v0); float den2 = 1.0f + p1;
        float p0 = 1.0f / den2;  p1 = p1 / den2;      // softmax over top-2
        // full softmax + logsumexp for aux losses
        float mx = acc[0];
        #pragma unroll
        for (int e = 1; e < Ec; ++e) mx = fmaxf(mx, acc[e]);
        float s = 0.0f;
        #pragma unroll
        for (int e = 0; e < Ec; ++e) s += expf(acc[e] - mx);
        float z = mx + logf(s);
        atomicAdd(zz_sum, z * z);
        #pragma unroll
        for (int e = 0; e < Ec; ++e) atomicAdd(&rp_sum[e], expf(acc[e] - mx) / s);
        atomicAdd(&assign_sum[i0], p0);
        atomicAdd(&assign_sum[i1], p1);
        topk_idx[2 * tok + 0] = i0;  topk_idx[2 * tok + 1] = i1;
        topk_probs[2 * tok + 0] = p0; topk_probs[2 * tok + 1] = p1;
        idx_out[4 * tok + 0] = i0; idx_out[4 * tok + 1] = 0;
        idx_out[4 * tok + 2] = i1; idx_out[4 * tok + 3] = 0;
    }
}

// ---------- dispatch: atomic capacity-limited slot assignment ----------
__global__ void dispatch_kernel(const int* __restrict__ topk_idx, int* cnt,
                                int* slot_map, int* kept, int* kept_total) {
    int i = blockIdx.x * blockDim.x + threadIdx.x;
    if (i >= NKs) return;
    int e = topk_idx[i];
    int pos = atomicAdd(&cnt[e], 1);
    int k = 0;
    if (pos < CAP) { slot_map[e * CAP + pos] = i; k = 1; }
    kept[i] = k;
    atomicAdd(kept_total, k);
}

__global__ void wslot_kernel(const float* __restrict__ topk_probs,
                             const int* __restrict__ kept, float* __restrict__ wslot) {
    int n = blockIdx.x * blockDim.x + threadIdx.x;
    if (n >= NTOK) return;
    float t0 = topk_probs[2 * n + 0] * (kept[2 * n + 0] ? 1.0f : 0.0f);
    float t1 = topk_probs[2 * n + 1] * (kept[2 * n + 1] ? 1.0f : 0.0f);
    float s = fmaxf(t0 + t1, 1e-12f);
    wslot[2 * n + 0] = t0 / s;
    wslot[2 * n + 1] = t1 / s;
}

// ---------- gather z = 2*x into expert-ordered bf16 rows ----------
__global__ __launch_bounds__(256) void gather_kernel(const int* __restrict__ slot_map,
                                                     const float* __restrict__ xf,
                                                     bf16_t* __restrict__ zb) {
    int row = blockIdx.x;                 // 0 .. E*CAP-1
    int slot = slot_map[row];
    bf16_t* dst = zb + (size_t)row * Dc;
    if (slot >= 0) {
        const float* src = xf + (size_t)(slot >> 1) * Dc;
        for (int d = threadIdx.x; d < Dc; d += blockDim.x) dst[d] = tobf(2.0f * src[d]);
    } else {
        for (int d = threadIdx.x; d < Dc; d += blockDim.x) dst[d] = tobf(0.0f);
    }
}

// ---------- expert FFN GEMM1 + SwiGLU ----------
__global__ __launch_bounds__(256) void ffn1_kernel(const bf16_t* __restrict__ zb,
                                                   const bf16_t* __restrict__ w13b,
                                                   bf16_t* __restrict__ hb) {
    int e = blockIdx.z;
    int lane = threadIdx.x & 31, wave = threadIdx.x >> 5;
    int m0 = blockIdx.y * 128 + wave * 16;
    int n0 = blockIdx.x * 64;
    const bf16_t* A  = zb   + (size_t)e * CAP * Dc;
    const bf16_t* Bg = w13b + (size_t)e * 2 * Hc * Dc;      // gate rows [0, H)
    const bf16_t* Bu = Bg + (size_t)Hc * Dc;                // up rows   [H, 2H)
    v8f accg[4], accu[4];
    #pragma unroll
    for (int t = 0; t < 4; ++t) { accg[t] = vzero(); accu[t] = vzero(); }
    gemm_body<4>(A + (size_t)m0 * Dc, Dc, Bg + (size_t)n0 * Dc, Dc, Dc, lane, accg);
    gemm_body<4>(A + (size_t)m0 * Dc, Dc, Bu + (size_t)n0 * Dc, Dc, Dc, lane, accu);
    int rowOff = (lane >> 4) * 8, col = lane & 15;
    bf16_t* hrowE = hb + (size_t)e * CAP * Hc;
    #pragma unroll
    for (int t = 0; t < 4; ++t) {
        #pragma unroll
        for (int i = 0; i < 8; ++i) {
            int m = m0 + rowOff + i;
            int n = n0 + t * 16 + col;
            float g = accg[t][i], u = accu[t][i];
            float h = (g / (1.0f + expf(-g))) * u;    // silu(g)*u
            hrowE[(size_t)m * Hc + n] = tobf(h);
        }
    }
}

// ---------- expert FFN GEMM2: y = seg + h @ w2^T, scatter to slot order ----------
__global__ __launch_bounds__(256) void ffn2_kernel(const bf16_t* __restrict__ hb,
                                                   const bf16_t* __restrict__ w2b,
                                                   const int* __restrict__ slot_map,
                                                   const float* __restrict__ xf,
                                                   float* __restrict__ sel_f,
                                                   bf16_t* __restrict__ selb) {
    int e = blockIdx.z;
    int lane = threadIdx.x & 31, wave = threadIdx.x >> 5;
    int m0 = blockIdx.y * 128 + wave * 16;
    int n0 = blockIdx.x * 64;
    const bf16_t* A = hb  + (size_t)e * CAP * Hc;
    const bf16_t* B = w2b + (size_t)e * Dc * Hc;
    v8f acc[4];
    #pragma unroll
    for (int t = 0; t < 4; ++t) acc[t] = vzero();
    gemm_body<4>(A + (size_t)m0 * Hc, Hc, B + (size_t)n0 * Hc, Hc, Hc, lane, acc);
    int rowOff = (lane >> 4) * 8, col = lane & 15;
    #pragma unroll
    for (int t = 0; t < 4; ++t) {
        #pragma unroll
        for (int i = 0; i < 8; ++i) {
            int m = m0 + rowOff + i;
            int n = n0 + t * 16 + col;
            int slot = slot_map[e * CAP + m];
            if (slot >= 0) {
                float v = acc[t][i] + xf[(size_t)(slot >> 1) * Dc + n];
                sel_f[(size_t)slot * Dc + n] = v;
                selb[(size_t)slot * Dc + n] = tobf(v);
            }
        }
    }
}

// ---------- plain GEMM: C = A @ B^T (f32 and/or bf16 out) ----------
__global__ __launch_bounds__(256) void gemm_nt_kernel(const bf16_t* __restrict__ A,
                                                      const bf16_t* __restrict__ B,
                                                      float* __restrict__ Cf,
                                                      bf16_t* __restrict__ Cb,
                                                      int K_, int lda, int ldb, int ldc) {
    int lane = threadIdx.x & 31, wave = threadIdx.x >> 5;
    int m0 = blockIdx.y * 128 + wave * 16;
    int n0 = blockIdx.x * 64;
    v8f acc[4];
    #pragma unroll
    for (int t = 0; t < 4; ++t) acc[t] = vzero();
    gemm_body<4>(A + (size_t)m0 * lda, lda, B + (size_t)n0 * ldb, ldb, K_, lane, acc);
    int rowOff = (lane >> 4) * 8, col = lane & 15;
    #pragma unroll
    for (int t = 0; t < 4; ++t) {
        #pragma unroll
        for (int i = 0; i < 8; ++i) {
            int m = m0 + rowOff + i;
            int n = n0 + t * 16 + col;
            float v = acc[t][i];
            if (Cf) Cf[(size_t)m * ldc + n] = v;
            if (Cb) Cb[(size_t)m * ldc + n] = tobf(v);
        }
    }
}

// ---------- GEMM + exact GELU epilogue ----------
__global__ __launch_bounds__(256) void gemm_gelu_kernel(const bf16_t* __restrict__ A,
                                                        const bf16_t* __restrict__ B,
                                                        bf16_t* __restrict__ Cb,
                                                        int K_, int lda, int ldb, int ldc) {
    int lane = threadIdx.x & 31, wave = threadIdx.x >> 5;
    int m0 = blockIdx.y * 128 + wave * 16;
    int n0 = blockIdx.x * 64;
    v8f acc[4];
    #pragma unroll
    for (int t = 0; t < 4; ++t) acc[t] = vzero();
    gemm_body<4>(A + (size_t)m0 * lda, lda, B + (size_t)n0 * ldb, ldb, K_, lane, acc);
    int rowOff = (lane >> 4) * 8, col = lane & 15;
    #pragma unroll
    for (int t = 0; t < 4; ++t) {
        #pragma unroll
        for (int i = 0; i < 8; ++i) {
            int m = m0 + rowOff + i;
            int n = n0 + t * 16 + col;
            float x = acc[t][i];
            float g = 0.5f * x * (1.0f + erff(x * 0.70710678118654752f));
            Cb[(size_t)m * ldc + n] = tobf(g);
        }
    }
}

// ---------- GEMM + refine epilogue: refined = (sel + acc) * km ----------
__global__ __launch_bounds__(256) void gemm_refine_kernel(const bf16_t* __restrict__ A,
                                                          const bf16_t* __restrict__ B,
                                                          const float* __restrict__ sel_f,
                                                          const int* __restrict__ kept,
                                                          float* __restrict__ refined,
                                                          int K_, int lda, int ldb) {
    int lane = threadIdx.x & 31, wave = threadIdx.x >> 5;
    int m0 = blockIdx.y * 128 + wave * 16;
    int n0 = blockIdx.x * 64;
    v8f acc[4];
    #pragma unroll
    for (int t = 0; t < 4; ++t) acc[t] = vzero();
    gemm_body<4>(A + (size_t)m0 * lda, lda, B + (size_t)n0 * ldb, ldb, K_, lane, acc);
    int rowOff = (lane >> 4) * 8, col = lane & 15;
    #pragma unroll
    for (int t = 0; t < 4; ++t) {
        #pragma unroll
        for (int i = 0; i < 8; ++i) {
            int m = m0 + rowOff + i;
            int n = n0 + t * 16 + col;
            float km = kept[m] ? 1.0f : 0.0f;
            refined[(size_t)m * Dc + n] = (sel_f[(size_t)m * Dc + n] + acc[t][i]) * km;
        }
    }
}

// ---------- K=2 collaboration attention: one wave per token ----------
__global__ __launch_bounds__(256) void collab_kernel(const float* __restrict__ Qf,
                                                     const float* __restrict__ Ktf,
                                                     const float* __restrict__ Mf,
                                                     const bf16_t* __restrict__ selb,
                                                     const int* __restrict__ kept,
                                                     bf16_t* __restrict__ catb) {
    int tok = (int)((blockIdx.x * (long long)blockDim.x + threadIdx.x) >> 5);
    if (tok >= NTOK) return;
    int lane = threadIdx.x & 31;
    float s00 = 0, s01 = 0, s10 = 0, s11 = 0;
    const size_t r0 = (size_t)(2 * tok) * Dc, r1 = (size_t)(2 * tok + 1) * Dc;
    for (int d = lane; d < Dc; d += 32) {
        float q0 = Qf[r0 + d], q1 = Qf[r1 + d];
        float k0 = Ktf[r0 + d], k1 = Ktf[r1 + d];
        s00 += q0 * k0; s01 += q0 * k1; s10 += q1 * k0; s11 += q1 * k1;
    }
    for (int off = 16; off > 0; off >>= 1) {
        s00 += __shfl_xor(s00, off, 32);
        s01 += __shfl_xor(s01, off, 32);
        s10 += __shfl_xor(s10, off, 32);
        s11 += __shfl_xor(s11, off, 32);
    }
    const float scale = 0.03608439182435161f;   // 1/sqrt(768)
    bool km0 = kept[2 * tok + 0] != 0, km1 = kept[2 * tok + 1] != 0;
    float Aat[2][2]; float sv[2][2] = {{s00, s01}, {s10, s11}};
    bool kmr[2] = {km0, km1};
    #pragma unroll
    for (int k = 0; k < 2; ++k) {
        float v0 = (kmr[k] && km0) ? sv[k][0] * scale : -3.0e38f;
        float v1 = (kmr[k] && km1) ? sv[k][1] * scale : -3.0e38f;
        float mx = fmaxf(v0, v1);
        float e0 = (kmr[k] && km0) ? expf(v0 - mx) : 0.0f;
        float e1 = (kmr[k] && km1) ? expf(v1 - mx) : 0.0f;
        float den = fmaxf(e0 + e1, 1e-12f);
        Aat[k][0] = e0 / den; Aat[k][1] = e1 / den;
    }
    bf16_t* c0 = catb + (size_t)(2 * tok) * D2;
    bf16_t* c1 = catb + (size_t)(2 * tok + 1) * D2;
    for (int d = lane; d < Dc; d += 32) {
        float m0v = Mf[r0 + d], m1v = Mf[r1 + d];
        c0[d] = selb[r0 + d];
        c1[d] = selb[r1 + d];
        c0[Dc + d] = tobf(Aat[0][0] * m0v + Aat[0][1] * m1v);
        c1[Dc + d] = tobf(Aat[1][0] * m0v + Aat[1][1] * m1v);
    }
}

// ---------- weighted fuse of the two slots per token ----------
__global__ void fuse_kernel(const float* __restrict__ refined,
                            const float* __restrict__ wslot,
                            bf16_t* __restrict__ fusedb) {
    long long i = (long long)blockIdx.x * blockDim.x + threadIdx.x;
    if (i >= (long long)NTOK * Dc) return;
    int n = (int)(i / Dc), d = (int)(i % Dc);
    float v = refined[(size_t)(2 * n) * Dc + d] * wslot[2 * n] +
              refined[(size_t)(2 * n + 1) * Dc + d] * wslot[2 * n + 1];
    fusedb[i] = tobf(v);
}

// ---------- aux scalar ----------
__global__ void finalize_aux_kernel(const float* rp_sum, const float* assign_sum,
                                    const float* zz_sum, const int* kept_total,
                                    float* aux_out) {
    if (threadIdx.x == 0 && blockIdx.x == 0) {
        float bal = 0.0f;
        for (int e = 0; e < Ec; ++e)
            bal += (rp_sum[e] / (float)NTOK) * (assign_sum[e] / (float)NTOK);
        bal *= (float)Ec;
        float zl = zz_sum[0] / (float)NTOK;
        float keep_mean = (float)kept_total[0] / (float)NKs;
        aux_out[0] = 0.01f * bal + 0.001f * zl + 0.001f * (1.0f - keep_mean);
    }
}

// ===================== host launcher =====================
extern "C" void kernel_launch(void* const* d_in, const int* in_sizes, int n_in,
                              void* d_out, int out_size, void* d_ws, size_t ws_size,
                              hipStream_t stream) {
    const float* xf   = (const float*)d_in[0];
    const float* gw   = (const float*)d_in[1];
    const float* w13  = (const float*)d_in[2];
    const float* w2   = (const float*)d_in[3];
    const float* msgw = (const float*)d_in[4];
    const float* qw   = (const float*)d_in[5];
    const float* kw   = (const float*)d_in[6];
    const float* ow   = (const float*)d_in[7];
    const float* u1w  = (const float*)d_in[8];
    const float* u2w  = (const float*)d_in[9];

    float* outF   = (float*)d_out;
    float* auxp   = outF + (size_t)NTOK * Dc;
    int*   idxOut = (int*)(auxp + 1);          // int64 indices as int pairs

    char* ws = (char*)d_ws;
    size_t off = 0;
    auto carve = [&](size_t bytes) -> char* {
        char* p = ws + off;
        off += (bytes + 255) & ~(size_t)255;
        return p;
    };
    bf16_t* w13b  = (bf16_t*)carve(sizeof(bf16_t) * (size_t)Ec * 2 * Hc * Dc);
    bf16_t* w2b   = (bf16_t*)carve(sizeof(bf16_t) * (size_t)Ec * Dc * Hc);
    bf16_t* msgb  = (bf16_t*)carve(sizeof(bf16_t) * (size_t)Dc * Dc);
    bf16_t* qb    = (bf16_t*)carve(sizeof(bf16_t) * (size_t)Dc * Dc);
    bf16_t* kb    = (bf16_t*)carve(sizeof(bf16_t) * (size_t)Dc * Dc);
    bf16_t* ob    = (bf16_t*)carve(sizeof(bf16_t) * (size_t)Dc * Dc);
    bf16_t* u1b   = (bf16_t*)carve(sizeof(bf16_t) * (size_t)D2 * D2);
    bf16_t* u2b   = (bf16_t*)carve(sizeof(bf16_t) * (size_t)Dc * D2);
    bf16_t* zb    = (bf16_t*)carve(sizeof(bf16_t) * (size_t)Ec * CAP * Dc);
    bf16_t* hb    = (bf16_t*)carve(sizeof(bf16_t) * (size_t)Ec * CAP * Hc);
    bf16_t* selb  = (bf16_t*)carve(sizeof(bf16_t) * (size_t)NKs * Dc);
    bf16_t* Mb    = (bf16_t*)carve(sizeof(bf16_t) * (size_t)NKs * Dc);
    bf16_t* catb  = (bf16_t*)carve(sizeof(bf16_t) * (size_t)NKs * D2);
    bf16_t* updhb = (bf16_t*)carve(sizeof(bf16_t) * (size_t)NKs * D2);
    bf16_t* fusedb= (bf16_t*)carve(sizeof(bf16_t) * (size_t)NTOK * Dc);
    float*  sel_f = (float*)carve(sizeof(float) * (size_t)NKs * Dc);
    float*  Mf    = (float*)carve(sizeof(float) * (size_t)NKs * Dc);
    float*  Qf    = (float*)carve(sizeof(float) * (size_t)NKs * Dc);
    float*  Ktf   = (float*)carve(sizeof(float) * (size_t)NKs * Dc);
    float*  reff  = (float*)carve(sizeof(float) * (size_t)NKs * Dc);
    float*  tkp   = (float*)carve(sizeof(float) * (size_t)NKs);
    float*  wslot = (float*)carve(sizeof(float) * (size_t)NKs);
    int*    tki   = (int*)carve(sizeof(int) * (size_t)NKs);
    int*    kept  = (int*)carve(sizeof(int) * (size_t)NKs);
    int*    smap  = (int*)carve(sizeof(int) * (size_t)Ec * CAP);
    float*  rp_sum     = (float*)carve(sizeof(float) * Ec);
    float*  assign_sum = (float*)carve(sizeof(float) * Ec);
    float*  zz_sum     = (float*)carve(sizeof(float) * 1);
    int*    cnt        = (int*)carve(sizeof(int) * Ec);
    int*    kept_total = (int*)carve(sizeof(int) * 1);

    auto cvt = [&](const float* s, bf16_t* d, long long n) {
        int blocks = (int)((n + 255) / 256);
        cvt_bf16_kernel<<<blocks, 256, 0, stream>>>(s, d, n);
    };

    // --- init scratch state ---
    fill_f32_kernel<<<1, 32, 0, stream>>>(rp_sum, 0.0f, Ec);
    fill_f32_kernel<<<1, 32, 0, stream>>>(assign_sum, 0.0f, Ec);
    fill_f32_kernel<<<1, 32, 0, stream>>>(zz_sum, 0.0f, 1);
    fill_i32_kernel<<<1, 32, 0, stream>>>(cnt, 0, Ec);
    fill_i32_kernel<<<1, 32, 0, stream>>>(kept_total, 0, 1);
    fill_i32_kernel<<<(Ec * CAP + 255) / 256, 256, 0, stream>>>(smap, -1, (long long)Ec * CAP);
    fill_f32_kernel<<<49152, 256, 0, stream>>>(sel_f, 0.0f, (long long)NKs * Dc);

    // --- weight conversion f32 -> bf16 ---
    cvt(w13, w13b, (long long)Ec * 2 * Hc * Dc);
    cvt(w2,  w2b,  (long long)Ec * Dc * Hc);
    cvt(msgw, msgb, (long long)Dc * Dc);
    cvt(qw,   qb,   (long long)Dc * Dc);
    cvt(kw,   kb,   (long long)Dc * Dc);
    cvt(ow,   ob,   (long long)Dc * Dc);
    cvt(u1w,  u1b,  (long long)D2 * D2);
    cvt(u2w,  u2b,  (long long)Dc * D2);

    // --- router / dispatch / gather ---
    router_kernel<<<NTOK / 8, 256, 0, stream>>>(xf, gw, tkp, tki, idxOut,
                                                rp_sum, assign_sum, zz_sum);
    dispatch_kernel<<<(NKs + 255) / 256, 256, 0, stream>>>(tki, cnt, smap, kept, kept_total);
    wslot_kernel<<<(NTOK + 255) / 256, 256, 0, stream>>>(tkp, kept, wslot);
    gather_kernel<<<Ec * CAP, 256, 0, stream>>>(smap, xf, zb);

    // --- expert FFN (WMMA) ---
    ffn1_kernel<<<dim3(Hc / 64, CAP / 128, Ec), 256, 0, stream>>>(zb, w13b, hb);
    ffn2_kernel<<<dim3(Dc / 64, CAP / 128, Ec), 256, 0, stream>>>(hb, w2b, smap, xf, sel_f, selb);

    // --- collaboration GEMMs (WMMA) ---
    gemm_nt_kernel<<<dim3(Dc / 64, NKs / 128), 256, 0, stream>>>(selb, msgb, Mf, Mb, Dc, Dc, Dc, Dc);
    gemm_nt_kernel<<<dim3(Dc / 64, NKs / 128), 256, 0, stream>>>(selb, qb, Qf, (bf16_t*)nullptr, Dc, Dc, Dc, Dc);
    gemm_nt_kernel<<<dim3(Dc / 64, NKs / 128), 256, 0, stream>>>(Mb, kb, Ktf, (bf16_t*)nullptr, Dc, Dc, Dc, Dc);

    collab_kernel<<<NTOK / 8, 256, 0, stream>>>(Qf, Ktf, Mf, selb, kept, catb);

    gemm_gelu_kernel<<<dim3(D2 / 64, NKs / 128), 256, 0, stream>>>(catb, u1b, updhb, D2, D2, D2, D2);
    gemm_refine_kernel<<<dim3(Dc / 64, NKs / 128), 256, 0, stream>>>(updhb, u2b, sel_f, kept, reff, D2, D2, D2);

    fuse_kernel<<<(int)(((long long)NTOK * Dc + 255) / 256), 256, 0, stream>>>(reff, wslot, fusedb);

    // --- output projection straight into d_out ---
    gemm_nt_kernel<<<dim3(Dc / 64, NTOK / 128), 256, 0, stream>>>(fusedb, ob, outF, (bf16_t*)nullptr, Dc, Dc, Dc, Dc);

    finalize_aux_kernel<<<1, 32, 0, stream>>>(rp_sum, assign_sum, zz_sum, kept_total, auxp);
}
